// GraphNetBlock_86346022519032
// MI455X (gfx1250) — compile-verified, matrix operand order checked by reading, not run on hardware
//
#include <hip/hip_runtime.h>
#include <cstdint>
#include <cstddef>

#define NN 50000
#define EE 200000
#define DD 256
#define HH 8
#define DHD 512

typedef __attribute__((ext_vector_type(16))) __bf16 v16bf;
typedef __attribute__((ext_vector_type(8)))  float  v8f;

union BF16Frag { unsigned int u[8]; __bf16 h[16]; v16bf v; };

// ---------------------------------------------------------------------------
// WMMA GEMM: out = epilogue(A[M,K] @ W[K,Nfull] + bias)
//   EPI 0: out[row,col] = acc + bias                    (out ld = Nfull)
//   EPI 1: ActGLU: accA = cols [0,Nhalf), accG = cols [Nhalf,Nfull)
//          out[row,col] = (accA+b_a) * relu(accG+b_g)   (out ld = Nhalf)
//   EPI 2: out[row,col] = res[row,col] + alpha * (acc + bias)
//
// Block = 128 threads = 4 waves. Block tile = 16 rows x 128 cols; each wave
// owns two 16-col slabs (cols wave*16 and wave*16+64), reusing one A fragment
// for 2 (EPI0/2) or 4 (EPI1) WMMAs per 32-deep K-step.
// A tile staged raw fp32 via CDNA5 async global->LDS DMA; W tiles staged via
// coalesced loads + native bf16 converts + packed ds_store_b64 ([n][k] LDS).
// Requires M % 16 == 0, K % 32 == 0, (EPI==1 ? Nfull/2 : Nfull) % 128 == 0.
// ---------------------------------------------------------------------------
template<int EPI>
__global__ __launch_bounds__(128)
void wmma_gemm(const float* __restrict__ A, const float* __restrict__ W,
               const float* __restrict__ bias, const float* __restrict__ res,
               const float* __restrict__ alpha, float* __restrict__ out,
               int M, int K, int Nfull)
{
    __shared__ float          lAf[16][32];      // raw fp32 A tile (async DMA dest)
    __shared__ unsigned short lB[2][128][36];   // bf16 bits, transposed [n][k], pad->b64 align

    const int tid  = threadIdx.x;
    const int wave = tid >> 5;
    const int lane = tid & 31;
    const int lm   = lane & 15;
    const int hi   = lane >> 4;

    const int rb    = blockIdx.x << 4;
    const int Nhalf = Nfull >> 1;
    const int cb    = blockIdx.y << 7;          // 128 cols per block

    v8f acc0[2] = {{}, {}};
    v8f acc1[2] = {{}, {}};

    // per-thread async A coords: one b128 (4 fp32) per thread per K-step
    const int ar  = tid >> 3;
    const int ac4 = (tid & 7) << 2;
    auto ldsA = (__attribute__((address_space(3))) const void*)&lAf[ar][ac4];

    for (int k0 = 0; k0 < K; k0 += 32) {
        // ---- A tile: async global->LDS copy (16x32 fp32 = 2KB, b128/lane) ----
        {
            unsigned long long ga = (unsigned long long)(const void*)
                (&A[(size_t)(rb + ar) * K + (k0 + ac4)]);
            asm volatile("global_load_async_to_lds_b128 %0, %1, off"
                         :: "v"(ldsA), "v"(ga) : "memory");
        }
        // ---- W tile(s): 32x128, coalesced loads, native cvt, b64 stores ----
        auto stageW = [&](int hx) {
            const float* Wp = W + (size_t)k0 * Nfull + cb + hx * Nhalf;
            const int n = tid;
#pragma unroll
            for (int i = 0; i < 8; ++i) {
                const int kk4 = i << 2;
                float g0 = Wp[(size_t)(kk4 + 0) * Nfull + n];
                float g1 = Wp[(size_t)(kk4 + 1) * Nfull + n];
                float g2 = Wp[(size_t)(kk4 + 2) * Nfull + n];
                float g3 = Wp[(size_t)(kk4 + 3) * Nfull + n];
                union { __bf16 h[4]; uint2 u2; } pk;
                pk.h[0] = (__bf16)g0;
                pk.h[1] = (__bf16)g1;
                pk.h[2] = (__bf16)g2;
                pk.h[3] = (__bf16)g3;
                *(uint2*)&lB[hx][n][kk4] = pk.u2;
            }
        };
        stageW(0);
        if (EPI == 1) stageW(1);
        if (tid == 0 && k0 + 32 < K)
            __builtin_prefetch(&W[(size_t)(k0 + 32) * Nfull + cb], 0, 1);
        asm volatile("s_wait_asynccnt 0x0" ::: "memory");
        __syncthreads();

        // ---- A fragment: native cvt of fp32 pairs from LDS (ISA 7.12.2) ----
        BF16Frag a;
#pragma unroll
        for (int j = 0; j < 8; ++j) {
            int i  = j << 1;
            int kk = (i & 7) + ((i >> 3) << 4) + (hi << 3);
            float2 f = *(const float2*)&lAf[lm][kk];
            a.h[i]     = (__bf16)f.x;
            a.h[i + 1] = (__bf16)f.y;
        }
        // ---- two column tiles per wave, reusing the A fragment ----
#pragma unroll
        for (int ct = 0; ct < 2; ++ct) {
            const int nloc = (wave << 4) + (ct << 6) + lm;
            BF16Frag b;
#pragma unroll
            for (int j = 0; j < 8; ++j) {
                int i  = j << 1;
                int kk = (i & 7) + ((i >> 3) << 4) + (hi << 3);
                b.u[j] = *(const unsigned int*)&lB[0][nloc][kk];
            }
            acc0[ct] = __builtin_amdgcn_wmma_f32_16x16x32_bf16(
                false, a.v, false, b.v, (short)0, acc0[ct], false, false);
            if (EPI == 1) {
                BF16Frag bg;
#pragma unroll
                for (int j = 0; j < 8; ++j) {
                    int i  = j << 1;
                    int kk = (i & 7) + ((i >> 3) << 4) + (hi << 3);
                    bg.u[j] = *(const unsigned int*)&lB[1][nloc][kk];
                }
                acc1[ct] = __builtin_amdgcn_wmma_f32_16x16x32_bf16(
                    false, a.v, false, bg.v, (short)0, acc1[ct], false, false);
            }
        }
        __syncthreads();
    }

    // ---- epilogue: VGPR r holds rows (r, r+8) split by lane half ----
    const int rbase = rb + (hi << 3);
#pragma unroll
    for (int ct = 0; ct < 2; ++ct) {
        const int col = cb + (wave << 4) + (ct << 6) + lm;
        if (EPI == 0) {
            float bv = bias[col];
#pragma unroll
            for (int r = 0; r < 8; ++r)
                out[(size_t)(rbase + r) * Nfull + col] = acc0[ct][r] + bv;
        } else if (EPI == 1) {
            float ba = bias[col];
            float bg = bias[col + Nhalf];
#pragma unroll
            for (int r = 0; r < 8; ++r) {
                float av = acc0[ct][r] + ba;
                float gv = acc1[ct][r] + bg;
                out[(size_t)(rbase + r) * Nhalf + col] = av * fmaxf(gv, 0.0f);
            }
        } else {
            float bv = bias[col];
            float al = alpha[0];
#pragma unroll
            for (int r = 0; r < 8; ++r) {
                size_t idx = (size_t)(rbase + r) * Nfull + col;
                out[idx] = res[idx] + al * (acc0[ct][r] + bv);
            }
        }
    }
}

// ---------------------------------------------------------------------------
// Small dense projection (Nout = 8): one thread per output element.
// ---------------------------------------------------------------------------
__global__ void proj_small(const float* __restrict__ X, const float* __restrict__ W,
                           const float* __restrict__ b, float* __restrict__ out,
                           int M, int K, int Nout)
{
    int t = blockIdx.x * blockDim.x + threadIdx.x;
    if (t >= M * Nout) return;
    int row = t / Nout, col = t % Nout;
    float s = b[col];
    for (int k = 0; k < K; ++k) s += X[(size_t)row * K + k] * W[(size_t)k * Nout + col];
    out[t] = s;
}

__global__ void fill_f32(float* __restrict__ p, float v, long n)
{
    long t = (long)blockIdx.x * blockDim.x + threadIdx.x;
    if (t < n) p[t] = v;
}

__device__ inline void atomic_max_f32(float* addr, float val)
{
    unsigned int* ua = (unsigned int*)addr;
    unsigned int old = *ua;
    while (__uint_as_float(old) < val) {
        unsigned int assumed = old;
        old = atomicCAS(ua, assumed, __float_as_uint(val));
        if (old == assumed) break;
    }
}

// scores per (edge, head): n2n = q[dst]*k[src], n2e = q[dst]*ek[e]  (qk_dim == 1)
__global__ void edge_scores(const float* __restrict__ q, const float* __restrict__ k,
                            const float* __restrict__ ek,
                            const int* __restrict__ src, const int* __restrict__ dst,
                            float* __restrict__ s1, float* __restrict__ s2, int EH)
{
    int t = blockIdx.x * blockDim.x + threadIdx.x;
    if (t >= EH) return;
    int e = t >> 3, h = t & 7;
    float qd = q[(size_t)dst[e] * HH + h];
    s1[t] = qd * k[(size_t)src[e] * HH + h];
    s2[t] = qd * ek[t];
}

__global__ void edge_segmax(const float* __restrict__ s1, const float* __restrict__ s2,
                            const int* __restrict__ dst,
                            float* __restrict__ m1, float* __restrict__ m2, int EH)
{
    int t = blockIdx.x * blockDim.x + threadIdx.x;
    if (t >= EH) return;
    int h = t & 7, d = dst[t >> 3];
    atomic_max_f32(&m1[(size_t)d * HH + h], s1[t]);
    atomic_max_f32(&m2[(size_t)d * HH + h], s2[t]);
}

__global__ void edge_expsum(const float* __restrict__ s1, const float* __restrict__ s2,
                            const float* __restrict__ m1, const float* __restrict__ m2,
                            const int* __restrict__ dst,
                            float* __restrict__ ex1, float* __restrict__ ex2,
                            float* __restrict__ z1, float* __restrict__ z2, int EH)
{
    int t = blockIdx.x * blockDim.x + threadIdx.x;
    if (t >= EH) return;
    int h = t & 7, d = dst[t >> 3];
    float e1 = expf(s1[t] - m1[(size_t)d * HH + h]);
    float e2 = expf(s2[t] - m2[(size_t)d * HH + h]);
    ex1[t] = e1; ex2[t] = e2;
    atomicAdd(&z1[(size_t)d * HH + h], e1);
    atomicAdd(&z2[(size_t)d * HH + h], e2);
}

// wv = (a_n2n + a_n2e)*v[src] + a_n2e*ev   (a_e2n == a_n2e per source bug)
__global__ void edge_aggregate(const float* __restrict__ ex1, const float* __restrict__ ex2,
                               const float* __restrict__ z1, const float* __restrict__ z2,
                               const float* __restrict__ v, const float* __restrict__ ev,
                               const int* __restrict__ src, const int* __restrict__ dst,
                               float* __restrict__ zbuf, int total)
{
    int t = blockIdx.x * blockDim.x + threadIdx.x;
    if (t >= total) return;
    int e = t >> 9;            // / 512
    int d = t & 511;
    int h = d >> 6;
    int dn = dst[e], sn = src[e];
    float a1 = ex1[(size_t)e * HH + h] / z1[(size_t)dn * HH + h];
    float a2 = ex2[(size_t)e * HH + h] / z2[(size_t)dn * HH + h];
    float w  = (a1 + a2) * v[(size_t)sn * DHD + d] + a2 * ev[(size_t)t];
    atomicAdd(&zbuf[(size_t)dn * DHD + d], w);
}

// in-place LayerNorm(512) -> relu, one block per row
__global__ __launch_bounds__(256)
void ln_relu(float* __restrict__ z, const float* __restrict__ gamma,
             const float* __restrict__ beta)
{
    __shared__ float s1[256], s2[256];
    int row = blockIdx.x, t = threadIdx.x;
    float x0 = z[(size_t)row * DHD + t];
    float x1 = z[(size_t)row * DHD + 256 + t];
    s1[t] = x0 + x1;
    s2[t] = x0 * x0 + x1 * x1;
    __syncthreads();
    for (int off = 128; off > 0; off >>= 1) {
        if (t < off) { s1[t] += s1[t + off]; s2[t] += s2[t + off]; }
        __syncthreads();
    }
    float mu   = s1[0] * (1.0f / 512.0f);
    float var  = s2[0] * (1.0f / 512.0f) - mu * mu;
    float rstd = rsqrtf(var + 1e-5f);
    float y0 = (x0 - mu) * rstd * gamma[t] + beta[t];
    float y1 = (x1 - mu) * rstd * gamma[256 + t] + beta[256 + t];
    z[(size_t)row * DHD + t]       = fmaxf(y0, 0.0f);
    z[(size_t)row * DHD + 256 + t] = fmaxf(y1, 0.0f);
}

// ---------------------------------------------------------------------------
extern "C" void kernel_launch(void* const* d_in, const int* in_sizes, int n_in,
                              void* d_out, int out_size, void* d_ws, size_t ws_size,
                              hipStream_t stream)
{
    (void)in_sizes; (void)n_in; (void)out_size; (void)ws_size;

    const float* n_in_p = (const float*)d_in[0];
    const float* e_in_p = (const float*)d_in[1];
    const int*   srcp   = (const int*)d_in[2];
    const int*   dstp   = (const int*)d_in[3];
    const float* nf_w1  = (const float*)d_in[4];
    const float* nf_b1  = (const float*)d_in[5];
    const float* nf_w2  = (const float*)d_in[6];
    const float* nf_b2  = (const float*)d_in[7];
    const float* ef_w1  = (const float*)d_in[8];
    const float* ef_b1  = (const float*)d_in[9];
    const float* ef_w2  = (const float*)d_in[10];
    const float* ef_b2  = (const float*)d_in[11];
    const float* nf2_w1 = (const float*)d_in[12];
    const float* nf2_b1 = (const float*)d_in[13];
    const float* nf2_w2 = (const float*)d_in[14];
    const float* nf2_b2 = (const float*)d_in[15];
    const float* ef2_w1 = (const float*)d_in[16];
    const float* ef2_b1 = (const float*)d_in[17];
    const float* ef2_w2 = (const float*)d_in[18];
    const float* ef2_b2 = (const float*)d_in[19];
    const float* q_w  = (const float*)d_in[20];
    const float* q_b  = (const float*)d_in[21];
    const float* k_w  = (const float*)d_in[22];
    const float* k_b  = (const float*)d_in[23];
    // d_in[24]/[25] = eq_w/eq_b: e2n score is computed-but-unused in the
    // reference (and a_e2n reuses n2e), so eq is dead — skip it.
    const float* ek_w = (const float*)d_in[26];
    const float* ek_b = (const float*)d_in[27];
    const float* v_w  = (const float*)d_in[28];
    const float* v_b  = (const float*)d_in[29];
    const float* ev_w = (const float*)d_in[30];
    const float* ev_b = (const float*)d_in[31];
    const float* mix_gamma = (const float*)d_in[32];
    const float* mix_beta  = (const float*)d_in[33];
    const float* mix_w     = (const float*)d_in[34];
    const float* mix_b     = (const float*)d_in[35];
    const float* alpha_n   = (const float*)d_in[36];
    const float* alpha_e   = (const float*)d_in[37];

    float* out_n = (float*)d_out;
    float* out_e = out_n + (size_t)NN * DD;

    // ---- workspace carve-up ----
    float* ws  = (float*)d_ws;
    size_t off = 0;
    auto take = [&](size_t nf) { float* p = ws + off; off += nf; return p; };
    float* r_actE = take((size_t)EE * DHD);   // ActGLU output (edges), reused for ff2
    float* r_actN = take((size_t)NN * DHD);   // ActGLU output (nodes), reused for ff2
    float* r_ev   = take((size_t)EE * DHD);
    float* r_v    = take((size_t)NN * DHD);
    float* r_zbuf = take((size_t)NN * DHD);
    float* r_n1   = take((size_t)NN * DD);
    float* r_n2   = take((size_t)NN * DD);
    float* r_e1   = take((size_t)EE * DD);
    float* r_q    = take((size_t)NN * HH);
    float* r_k    = take((size_t)NN * HH);
    float* r_ek   = take((size_t)EE * HH);
    float* r_s1   = take((size_t)EE * HH);
    float* r_s2   = take((size_t)EE * HH);
    float* r_ex1  = take((size_t)EE * HH);
    float* r_ex2  = take((size_t)EE * HH);
    float* r_m1   = take((size_t)NN * HH);
    float* r_m2   = take((size_t)NN * HH);
    float* r_z1   = take((size_t)NN * HH);
    float* r_z2   = take((size_t)NN * HH);

    const dim3 blk(128);
    const int gN = NN / 16;     // 3125
    const int gE = EE / 16;     // 12500

    // ---- pre-FFNs with fused ActGLU + ReZero residual ----
    wmma_gemm<1><<<dim3(gN, 4), blk, 0, stream>>>(n_in_p, nf_w1, nf_b1, nullptr, nullptr,
                                                  r_actN, NN, DD, 4 * DD);
    wmma_gemm<2><<<dim3(gN, 2), blk, 0, stream>>>(r_actN, nf_w2, nf_b2, n_in_p, alpha_n,
                                                  r_n1, NN, 2 * DD, DD);
    wmma_gemm<1><<<dim3(gE, 4), blk, 0, stream>>>(e_in_p, ef_w1, ef_b1, nullptr, nullptr,
                                                  r_actE, EE, DD, 4 * DD);
    wmma_gemm<2><<<dim3(gE, 2), blk, 0, stream>>>(r_actE, ef_w2, ef_b2, e_in_p, alpha_e,
                                                  r_e1, EE, 2 * DD, DD);

    // ---- projections ----
    proj_small<<<(NN * HH + 255) / 256, 256, 0, stream>>>(r_n1, q_w, q_b, r_q, NN, DD, HH);
    proj_small<<<(NN * HH + 255) / 256, 256, 0, stream>>>(r_n1, k_w, k_b, r_k, NN, DD, HH);
    proj_small<<<(EE * HH + 255) / 256, 256, 0, stream>>>(r_e1, ek_w, ek_b, r_ek, EE, DD, HH);
    wmma_gemm<0><<<dim3(gN, 4), blk, 0, stream>>>(r_n1, v_w, v_b, nullptr, nullptr,
                                                  r_v, NN, DD, DHD);
    wmma_gemm<0><<<dim3(gE, 4), blk, 0, stream>>>(r_e1, ev_w, ev_b, nullptr, nullptr,
                                                  r_ev, EE, DD, DHD);

    // ---- edge softmax + aggregation (accumulators re-zeroed every call) ----
    fill_f32<<<(NN * HH + 255) / 256, 256, 0, stream>>>(r_m1, -3.0e38f, (long)NN * HH);
    fill_f32<<<(NN * HH + 255) / 256, 256, 0, stream>>>(r_m2, -3.0e38f, (long)NN * HH);
    fill_f32<<<(NN * HH + 255) / 256, 256, 0, stream>>>(r_z1, 0.0f, (long)NN * HH);
    fill_f32<<<(NN * HH + 255) / 256, 256, 0, stream>>>(r_z2, 0.0f, (long)NN * HH);
    fill_f32<<<((long)NN * DHD + 255) / 256, 256, 0, stream>>>(r_zbuf, 0.0f, (long)NN * DHD);

    edge_scores<<<(EE * HH + 255) / 256, 256, 0, stream>>>(r_q, r_k, r_ek, srcp, dstp,
                                                           r_s1, r_s2, EE * HH);
    edge_segmax<<<(EE * HH + 255) / 256, 256, 0, stream>>>(r_s1, r_s2, dstp, r_m1, r_m2,
                                                           EE * HH);
    edge_expsum<<<(EE * HH + 255) / 256, 256, 0, stream>>>(r_s1, r_s2, r_m1, r_m2, dstp,
                                                           r_ex1, r_ex2, r_z1, r_z2, EE * HH);
    edge_aggregate<<<((long)EE * DHD + 255) / 256, 256, 0, stream>>>(
        r_ex1, r_ex2, r_z1, r_z2, r_v, r_ev, srcp, dstp, r_zbuf, EE * DHD);

    // ---- LayerNorm -> relu -> mix GEMM with fused residual ----
    ln_relu<<<NN, 256, 0, stream>>>(r_zbuf, mix_gamma, mix_beta);
    wmma_gemm<2><<<dim3(gN, 2), blk, 0, stream>>>(r_zbuf, mix_w, mix_b, r_n1, alpha_n,
                                                  r_n2, NN, DHD, DD);

    // ---- post-FFNs, writing final outputs directly ----
    wmma_gemm<1><<<dim3(gN, 4), blk, 0, stream>>>(r_n2, nf2_w1, nf2_b1, nullptr, nullptr,
                                                  r_actN, NN, DD, 4 * DD);
    wmma_gemm<2><<<dim3(gN, 2), blk, 0, stream>>>(r_actN, nf2_w2, nf2_b2, r_n2, alpha_n,
                                                  out_n, NN, 2 * DD, DD);
    wmma_gemm<1><<<dim3(gE, 4), blk, 0, stream>>>(r_e1, ef2_w1, ef2_b1, nullptr, nullptr,
                                                  r_actE, EE, DD, 4 * DD);
    wmma_gemm<2><<<dim3(gE, 2), blk, 0, stream>>>(r_actE, ef2_w2, ef2_b2, r_e1, alpha_e,
                                                  out_e, EE, 2 * DD, DD);
}
